// ReconstructedANI_80032420593973
// MI455X (gfx1250) — compile-verified, matrix operand order checked by reading, not run on hardware
//
#include <hip/hip_runtime.h>
#include <hip/hip_bf16.h>

typedef __attribute__((ext_vector_type(16))) _Float16 v16h;
typedef __attribute__((ext_vector_type(8)))  _Float16 v8h;
typedef __attribute__((ext_vector_type(8)))  float    v8f;

#define NSPEC 7
#define RCUT  5.2f
#define PI_F  3.14159265358979323846f
#define MT    4   // M-tiles (16 rows each) per wave: 64x16 output per wave32

// ---------------------------------------------------------------------------
// AEV kernel: one block (128 threads) per atom i. LDS accumulator for the 112
// radial features, ds_add_f32 atomics, write f16 row padded to K=128.
// ---------------------------------------------------------------------------
__global__ void aev_kernel(const int* __restrict__ species,
                           const float* __restrict__ coords,
                           const float* __restrict__ EtaR,
                           const float* __restrict__ ShfR,
                           _Float16* __restrict__ aev_h, // [N,128]
                           int N) {
  __shared__ float acc[112];
  __shared__ float shf[16];
  const int i = blockIdx.x;
  const int t = threadIdx.x;
  if (t < 112) acc[t] = 0.0f;
  if (t < 16)  shf[t] = ShfR[t];
  __syncthreads();

  const float eta = EtaR[0];
  const float xi = coords[i * 3 + 0];
  const float yi = coords[i * 3 + 1];
  const float zi = coords[i * 3 + 2];

  for (int j = t; j < N; j += blockDim.x) {
    const float dx = xi - coords[j * 3 + 0];
    const float dy = yi - coords[j * 3 + 1];
    const float dz = zi - coords[j * 3 + 2];
    const float dsq = dx * dx + dy * dy + dz * dz;
    if (dsq <= 0.0f) continue;          // excludes self and coincident atoms
    const float dist = sqrtf(dsq);
    if (dist > RCUT) continue;          // cutoff mask
    const float fc = 0.5f * __cosf(PI_F * dist / RCUT) + 0.5f;
    const int base = species[j] * 16;
#pragma unroll
    for (int r = 0; r < 16; ++r) {
      const float d = dist - shf[r];
      atomicAdd(&acc[base + r], __expf(-eta * d * d) * fc);
    }
  }
  __syncthreads();
  for (int c = t; c < 128; c += blockDim.x)
    aev_h[(size_t)i * 128 + c] = (c < 112) ? (_Float16)acc[c] : (_Float16)0.0f;
}

// ---------------------------------------------------------------------------
// Weight conversion: crop W0 [7*256,1008] -> f16 [7*256,128] (112 live cols)
// ---------------------------------------------------------------------------
__global__ void cvt_w0_kernel(const float* __restrict__ src,
                              _Float16* __restrict__ dst,
                              int rows, int srcK, int dstK, int useK) {
  const int idx = blockIdx.x * blockDim.x + threadIdx.x;
  if (idx >= rows * dstK) return;
  const int c = idx % dstK;
  const int r = idx / dstK;
  dst[idx] = (c < useK) ? (_Float16)src[(size_t)r * srcK + c] : (_Float16)0.0f;
}

__global__ void cvt_f16_kernel(const float* __restrict__ src,
                               _Float16* __restrict__ dst, int n) {
  const int idx = blockIdx.x * blockDim.x + threadIdx.x;
  if (idx < n) dst[idx] = (_Float16)src[idx];
}

// ---------------------------------------------------------------------------
// WMMA fragment load: 16-bit A (16x32 MxK) / B (32x16 KxN from [N,K] rows).
// Lane L reads row/col (L&15), two 16B chunks at K offsets (L>>4)*8 and +16.
// ---------------------------------------------------------------------------
__device__ inline v16h load_frag16(const _Float16* __restrict__ base,
                                   int ld, int rc0, int k, int lane) {
  const int half = lane >> 4;
  const int idx  = lane & 15;
  const _Float16* p = base + (size_t)(rc0 + idx) * ld + k + half * 8;
  const v8h lo = *(const v8h*)(p);
  const v8h hi = *(const v8h*)(p + 16);
  v16h r;
#pragma unroll
  for (int e = 0; e < 8; ++e) { r[e] = lo[e]; r[e + 8] = hi[e]; }
  return r;
}

__device__ inline float celu01(float x) {
  // celu(x, alpha=0.1) = max(x,0) + min(0, 0.1*expm1(x/0.1))
  return (x > 0.0f) ? x : 0.1f * expm1f(x * 10.0f);
}

// ---------------------------------------------------------------------------
// GEMM + bias + CELU. Each wave32 computes a 64x16 output block (MT=4 tiles of
// 16x16), reusing one B (weight) fragment across 4 WMMAs per K-step:
//   2.5 b128 loads per v_wmma instead of 4.
//   Out[s][m,n] = celu( sum_k A[s][m,k] * W[s][n,k] + bias[s][n] )
// aSpStride==0 => A shared across species (layer 0: the AEV).
// All control flow is wave-uniform -> EXEC all-ones around v_wmma.
// ---------------------------------------------------------------------------
__global__ void gemm_celu_kernel(const _Float16* __restrict__ A, size_t aSpStride,
                                 const _Float16* __restrict__ W,
                                 const float* __restrict__ bias, int biasStride,
                                 _Float16* __restrict__ Out,
                                 int M, int N, int K, int doCelu) {
  const int lane = threadIdx.x & 31;
  const int wave = threadIdx.x >> 5;
  const int tilesMG = M >> 6;            // groups of 64 rows (MT*16)
  const int tilesN  = N >> 4;
  const int tilesPerS = tilesMG * tilesN;
  const int gt = blockIdx.x * (blockDim.x >> 5) + wave;
  if (gt >= NSPEC * tilesPerS) return;   // wave-uniform
  const int s  = gt / tilesPerS;
  const int t  = gt % tilesPerS;
  const int tm = (t % tilesMG) << 6;
  const int tn = (t / tilesMG) << 4;

  const _Float16* As = A + (size_t)s * aSpStride;
  const _Float16* Ws = W + (size_t)s * (size_t)N * K;

  v8f acc[MT];
#pragma unroll
  for (int mt = 0; mt < MT; ++mt) acc[mt] = (v8f){};

  for (int k = 0; k < K; k += 32) {
    const v16h b = load_frag16(Ws, K, tn, k, lane);   // shared across MT tiles
#pragma unroll
    for (int mt = 0; mt < MT; ++mt) {
      const v16h a = load_frag16(As, K, tm + mt * 16, k, lane);
      acc[mt] = __builtin_amdgcn_wmma_f32_16x16x32_f16(
          /*neg_a=*/false, a, /*neg_b=*/false, b,
          /*c_mod=*/(short)0, acc[mt], /*reuse_a=*/false, /*reuse_b=*/false);
    }
  }

  // Epilogue: C/D layout -> VGPR v holds (row base + v + 8*(lane>>4), col tn+(lane&15))
  const int col     = lane & 15;
  const int rowHalf = (lane >> 4) << 3;
  const float bcol  = bias[s * biasStride + tn + col];
  _Float16* Os = Out + (size_t)s * (size_t)M * N;
#pragma unroll
  for (int mt = 0; mt < MT; ++mt) {
    const int rowBase = tm + mt * 16 + rowHalf;
#pragma unroll
    for (int v = 0; v < 8; ++v) {
      float x = acc[mt][v] + bcol;
      if (doCelu) x = celu01(x);
      Os[(size_t)(rowBase + v) * N + (tn + col)] = (_Float16)x;
    }
  }
}

// ---------------------------------------------------------------------------
// Final layer: per atom n, e = bf[s] + dot(H2[s][n][0:160], Wf[s][0:160]) with
// s = species[n]; wave32 shuffle reduce, one global f32 atomic per atom.
// ---------------------------------------------------------------------------
__global__ void final_kernel(const _Float16* __restrict__ H2,
                             const float* __restrict__ Wf,
                             const float* __restrict__ bf,
                             const int* __restrict__ species,
                             float* __restrict__ out, int M) {
  const int lane = threadIdx.x & 31;
  const int wave = threadIdx.x >> 5;
  const int n = blockIdx.x * (blockDim.x >> 5) + wave;
  if (n >= M) return;
  const int s = species[n];
  const _Float16* h = H2 + ((size_t)s * M + n) * 160;
  const float* w = Wf + (size_t)s * 160;
  float acc = 0.0f;
  for (int k = lane; k < 160; k += 32) acc += (float)h[k] * w[k];
#pragma unroll
  for (int off = 16; off > 0; off >>= 1) acc += __shfl_down(acc, off, 32);
  if (lane == 0) atomicAdd(out, acc + bf[s]);
}

// ---------------------------------------------------------------------------
static inline size_t align256(size_t x) { return (x + 255) & ~(size_t)255; }

extern "C" void kernel_launch(void* const* d_in, const int* in_sizes, int n_in,
                              void* d_out, int out_size, void* d_ws, size_t ws_size,
                              hipStream_t stream) {
  const int*   species = (const int*)d_in[0];
  const float* coords  = (const float*)d_in[1];
  const float* EtaR    = (const float*)d_in[2];
  const float* ShfR    = (const float*)d_in[3];
  const float* W0 = (const float*)d_in[4];
  const float* b0 = (const float*)d_in[5];
  const float* W1 = (const float*)d_in[6];
  const float* b1 = (const float*)d_in[7];
  const float* W2 = (const float*)d_in[8];
  const float* b2 = (const float*)d_in[9];
  const float* Wf = (const float*)d_in[10];
  const float* bf = (const float*)d_in[11];
  float* out = (float*)d_out;

  const int M  = in_sizes[0];   // atoms (B*N = 2048)
  const int K0 = 128;           // padded live-AEV width (112 -> 128)
  const int N0 = 256, N1 = 192, N2 = 160;

  // Workspace carve-up (f16 regions, 256B-aligned)
  char* ws = (char*)d_ws;
  size_t off = 0;
  _Float16* aev_h = (_Float16*)(ws + off); off = align256(off + (size_t)M * K0 * 2);
  _Float16* W0h   = (_Float16*)(ws + off); off = align256(off + (size_t)NSPEC * N0 * K0 * 2);
  _Float16* W1h   = (_Float16*)(ws + off); off = align256(off + (size_t)NSPEC * N1 * N0 * 2);
  _Float16* W2h   = (_Float16*)(ws + off); off = align256(off + (size_t)NSPEC * N2 * N1 * 2);
  _Float16* H0    = (_Float16*)(ws + off); off = align256(off + (size_t)NSPEC * M * N0 * 2);
  _Float16* H1    = (_Float16*)(ws + off); off = align256(off + (size_t)NSPEC * M * N1 * 2);
  _Float16* H2    = (_Float16*)(ws + off); off = align256(off + (size_t)NSPEC * M * N2 * 2);
  (void)ws_size; (void)n_in; (void)out_size;

  hipMemsetAsync(out, 0, sizeof(float), stream);

  // 1) AEV features (VALU/trans-bound phase)
  aev_kernel<<<M, 128, 0, stream>>>(species, coords, EtaR, ShfR, aev_h, M);

  // 2) Weight conversions to f16 (L2-resident, ~4.8 MB total)
  {
    int n0 = NSPEC * N0 * K0;
    cvt_w0_kernel<<<(n0 + 255) / 256, 256, 0, stream>>>(W0, W0h, NSPEC * N0, 1008, K0, 112);
    int n1 = NSPEC * N1 * N0;
    cvt_f16_kernel<<<(n1 + 255) / 256, 256, 0, stream>>>(W1, W1h, n1);
    int n2 = NSPEC * N2 * N1;
    cvt_f16_kernel<<<(n2 + 255) / 256, 256, 0, stream>>>(W2, W2h, n2);
  }

  // 3) WMMA MLP layers (f16 in, f32 accumulate, fused bias+CELU)
  const int wavesPerBlock = 8; // 256 threads
  {
    int waves = NSPEC * (M >> 6) * (N0 >> 4);
    gemm_celu_kernel<<<(waves + wavesPerBlock - 1) / wavesPerBlock, 256, 0, stream>>>(
        aev_h, /*aSpStride=*/0, W0h, b0, N0, H0, M, N0, K0, 1);
  }
  {
    int waves = NSPEC * (M >> 6) * (N1 >> 4);
    gemm_celu_kernel<<<(waves + wavesPerBlock - 1) / wavesPerBlock, 256, 0, stream>>>(
        H0, (size_t)M * N0, W1h, b1, N1, H1, M, N1, N0, 1);
  }
  {
    int waves = NSPEC * (M >> 6) * (N2 >> 4);
    gemm_celu_kernel<<<(waves + wavesPerBlock - 1) / wavesPerBlock, 256, 0, stream>>>(
        H1, (size_t)M * N1, W2h, b2, N2, H2, M, N2, N1, 1);
  }

  // 4) Final 160-dot per atom, species-gathered, atomic accumulate
  final_kernel<<<(M + wavesPerBlock - 1) / wavesPerBlock, 256, 0, stream>>>(
      H2, Wf, bf, species, out, M);
}
